// TE_GNN_26628797235672
// MI455X (gfx1250) — matrix operand order repaired and checked by reading, not compiled, not run on hardware
//
#include <hip/hip_runtime.h>
#include <hip/hip_bf16.h>

// ---------------------------------------------------------------------------
// TE-GNN pipeline for MI455X (gfx1250, wave32, WMMA bf16 16x16x32).
// ---------------------------------------------------------------------------

typedef __attribute__((ext_vector_type(16))) __bf16 v16bf;
typedef __attribute__((ext_vector_type(8)))  float  v8f;
typedef __attribute__((ext_vector_type(8)))  unsigned short v8us;

__device__ __forceinline__ unsigned short f2bf_u(float f) {
  return __builtin_bit_cast(unsigned short, (__bf16)f);   // hw cvt, not int trick
}
__device__ __forceinline__ __bf16 us2bf(unsigned short h) {
  return __builtin_bit_cast(__bf16, h);
}
__device__ __forceinline__ float sigm(float x) { return 1.f / (1.f + __expf(-x)); }

// ---------------------------------------------------------------------------
// WMMA GEMM:  C[M,N] = A[M,K] * W[N,K]^T (+ bias[N]);  fp32 memory, bf16 math.
// K compile-time (=128 here) -> fully unrolled, 4 WMMAs per tile.
// Each wave: 1 M-tile x NT N-tiles, A fragments loaded once and reused.
// M % 16 == 0, N % (16*NT) == 0.
// revL > 0: A row (b*revL + l) read as (b*revL + revL-1-l) (time reversal).
// Tile-group id is wave-uniform so the early-return keeps EXEC all-ones
// around the WMMA ops (ISA requirement).
// ---------------------------------------------------------------------------
template <int K, int NT>
__global__ void k_gemm_f32_wmma(const float* __restrict__ A,
                                const float* __restrict__ W,
                                const float* __restrict__ bias,
                                float* __restrict__ C,
                                int M, int N, int revL)
{
  int lane = threadIdx.x & 31;
  int wave = (int)((blockIdx.x * (unsigned)blockDim.x + threadIdx.x) >> 5);
  int NG = N / (16 * NT);
  int total = (M >> 4) * NG;
  if (wave >= total) return;
  int mt = wave / NG, ng = wave % NG;
  int half = lane >> 4;        // K sub-range select within fragment
  int lid  = lane & 15;        // A row / B col within tile

  int arow = mt * 16 + lid;
  if (revL > 0) {              // time-reversed read of A rows
    int b = arow / revL, l = arow % revL;
    arow = b * revL + (revL - 1 - l);
  }
  const float* ap = A + (size_t)arow * K;

  // load A fragments once (ISA 16-bit A 16x32 layout: lanes 0-15 K 0-7/16-23,
  // lanes 16-31 K 8-15/24-31), reuse across the NT B tiles
  v16bf afrag[K / 32];
#pragma unroll
  for (int kk = 0; kk < K / 32; ++kk) {
    const float* pa0 = ap + kk * 32 + half * 8;
    const float* pa1 = pa0 + 16;
#pragma unroll
    for (int i = 0; i < 8; ++i) {
      afrag[kk][i]     = (__bf16)pa0[i];
      afrag[kk][8 + i] = (__bf16)pa1[i];
    }
  }

  v8f zero = {};
  v8f acc[NT];
#pragma unroll
  for (int j = 0; j < NT; ++j) acc[j] = zero;

#pragma unroll
  for (int j = 0; j < NT; ++j) {
    int ncol = (ng * NT + j) * 16 + lid;
    const float* wp = W + (size_t)ncol * K;
#pragma unroll
    for (int kk = 0; kk < K / 32; ++kk) {
      v16bf b;                                  // B[k][n] = W[n][k], K-contig
      const float* pb = wp + kk * 32 + half * 8;
#pragma unroll
      for (int i = 0; i < 16; ++i) b[i] = (__bf16)pb[i < 8 ? i : i + 8];
      acc[j] = __builtin_amdgcn_wmma_f32_16x16x32_bf16(false, afrag[kk], false, b,
                                                       (short)0, acc[j], false, false);
    }
  }

#pragma unroll
  for (int j = 0; j < NT; ++j) {
    int ncol = (ng * NT + j) * 16 + lid;
    float bn = bias ? bias[ncol] : 0.f;
#pragma unroll
    for (int r = 0; r < 8; ++r) {
      int m = mt * 16 + half * 8 + r;           // f32 C/D layout: rows r, r+8
      C[(size_t)m * N + ncol] = acc[j][r] + bn;
    }
  }
}

// ---------------------------------------------------------------------------
// WMMA GEMM, both operands pre-converted bf16 (final scoring GEMM).
// Each wave: 1 M-tile x NT N-tiles; z_final A fragment reused across tiles.
// Store only rows m < Mreal (M padded to 16).
// ---------------------------------------------------------------------------
template <int K, int NT>
__global__ void k_gemm_bf16_wmma(const unsigned short* __restrict__ A,
                                 const unsigned short* __restrict__ Bm,
                                 float* __restrict__ C,
                                 int Mpad, int Mreal, int N)
{
  int lane = threadIdx.x & 31;
  int wave = (int)((blockIdx.x * (unsigned)blockDim.x + threadIdx.x) >> 5);
  int NG = N / (16 * NT);
  int total = (Mpad >> 4) * NG;
  if (wave >= total) return;
  int mt = wave / NG, ng = wave % NG;
  int half = lane >> 4;
  int lid  = lane & 15;

  const unsigned short* ap = A + (size_t)(mt * 16 + lid) * K;
  v16bf afrag[K / 32];
#pragma unroll
  for (int kk = 0; kk < K / 32; ++kk) {
    v8us a0 = *(const v8us*)(ap + kk * 32 + half * 8);
    v8us a1 = *(const v8us*)(ap + kk * 32 + 16 + half * 8);
#pragma unroll
    for (int i = 0; i < 8; ++i) {
      afrag[kk][i]     = us2bf(a0[i]);
      afrag[kk][8 + i] = us2bf(a1[i]);
    }
  }

  v8f zero = {};
  v8f acc[NT];
#pragma unroll
  for (int j = 0; j < NT; ++j) acc[j] = zero;

#pragma unroll
  for (int j = 0; j < NT; ++j) {
    const unsigned short* bp = Bm + (size_t)((ng * NT + j) * 16 + lid) * K;
#pragma unroll
    for (int kk = 0; kk < K / 32; ++kk) {
      v8us b0 = *(const v8us*)(bp + kk * 32 + half * 8);
      v8us b1 = *(const v8us*)(bp + kk * 32 + 16 + half * 8);
      v16bf b;
#pragma unroll
      for (int i = 0; i < 8; ++i) { b[i] = us2bf(b0[i]); b[8 + i] = us2bf(b1[i]); }
      acc[j] = __builtin_amdgcn_wmma_f32_16x16x32_bf16(false, afrag[kk], false, b,
                                                       (short)0, acc[j], false, false);
    }
  }

#pragma unroll
  for (int j = 0; j < NT; ++j) {
    int ncol = (ng * NT + j) * 16 + lid;
#pragma unroll
    for (int r = 0; r < 8; ++r) {
      int m = mt * 16 + half * 8 + r;
      if (m < Mreal) C[(size_t)m * N + ncol] = acc[j][r];
    }
  }
}

// ---------------------------------------------------------------------------
// Elementwise / gather / scatter helpers
// ---------------------------------------------------------------------------
__global__ void k_zero(float* __restrict__ p, long n) {
  long i = blockIdx.x * (long)blockDim.x + threadIdx.x;
  if (i < n) p[i] = 0.f;
}

// acc = x = item_emb[items]
__global__ void k_gather_items(const int* __restrict__ items,
                               const float* __restrict__ emb,
                               float* __restrict__ acc, float* __restrict__ x,
                               int T, int D) {
  long i = blockIdx.x * (long)blockDim.x + threadIdx.x;
  if (i >= (long)T * D) return;
  int t = (int)(i / D), d = (int)(i % D);
  float v = emb[(size_t)items[t] * D + d];
  acc[i] = v; x[i] = v;
}

__global__ void k_gcn_scatter(const int* __restrict__ src, const int* __restrict__ dst,
                              const float* __restrict__ ew,
                              const float* __restrict__ x, float* __restrict__ xnew,
                              int E, int D) {
  long i = blockIdx.x * (long)blockDim.x + threadIdx.x;
  if (i >= (long)E * D) return;
  int e = (int)(i / D), d = (int)(i % D);
  atomicAdd(&xnew[(size_t)dst[e] * D + d], x[(size_t)src[e] * D + d] * ew[e]);
}

__global__ void k_gcn_accum(float* __restrict__ acc, float* __restrict__ x,
                            const float* __restrict__ xnew, long n) {
  long i = blockIdx.x * (long)blockDim.x + threadIdx.x;
  if (i >= n) return;
  float v = xnew[i];
  acc[i] += v;
  x[i] = v;
}

__global__ void k_hfinal(const float* __restrict__ acc, const int* __restrict__ idx,
                         float* __restrict__ hfin, int T, int D, float scale) {
  long i = blockIdx.x * (long)blockDim.x + threadIdx.x;
  if (i >= (long)T * D) return;
  int t = (int)(i / D), d = (int)(i % D);
  hfin[i] = acc[(size_t)idx[t] * D + d] * scale;
}

__global__ void k_bin_gather(const int* __restrict__ bins, const float* __restrict__ emb,
                             float* __restrict__ e, int NT, int D) {
  long i = blockIdx.x * (long)blockDim.x + threadIdx.x;
  if (i >= (long)NT * D) return;
  int t = (int)(i / D), d = (int)(i % D);
  e[i] = emb[(size_t)bins[t] * D + d];
}

// One fused GRU step (gates r,z,n as in torch).
// xp includes bih (from projection GEMM); gh includes bhh.
// out == nullptr for the "short" GRU (keep only final hidden state).
__global__ void k_gru_gate(const float* __restrict__ xp, const float* __restrict__ gh,
                           float* __restrict__ h, float* __restrict__ out,
                           int t_in, int t_out, int B, int L, int D) {
  long i = blockIdx.x * (long)blockDim.x + threadIdx.x;
  if (i >= (long)B * D) return;
  int b = (int)(i / D), d = (int)(i % D);
  const float* xpt = xp + ((size_t)b * L + t_in) * (size_t)(3 * D);
  const float* ghb = gh + (size_t)b * (3 * D);
  float r = sigm(xpt[d] + ghb[d]);
  float z = sigm(xpt[D + d] + ghb[D + d]);
  float n = tanhf(xpt[2 * D + d] + r * ghb[2 * D + d]);
  float hp = h[(size_t)b * D + d];
  float hn = (1.f - z) * n + z * hp;
  h[(size_t)b * D + d] = hn;
  if (out) out[((size_t)b * L + t_out) * D + d] = hn;
}

// b_gate = sigmoid([out_f|out_b] . w1); e_hat_full written in place over out_f.
__global__ void k_egate(float* __restrict__ outf, const float* __restrict__ outb,
                        const float* __restrict__ w1, int D) {
  int tok = blockIdx.x, d = threadIdx.x;
  float f  = outf[(size_t)tok * D + d];
  float bb = outb[(size_t)tok * D + d];
  __shared__ float s[128];
  s[d] = f * w1[d] + bb * w1[D + d];
  __syncthreads();
  for (int o = blockDim.x >> 1; o > 0; o >>= 1) {
    if (d < o) s[d] += s[d + o];
    __syncthreads();
  }
  float g = sigm(s[0]);
  outf[(size_t)tok * D + d] = g * f + (1.f - g) * bb;
}

__global__ void k_gather_ehat(const float* __restrict__ full, const int* __restrict__ seg,
                              const int* __restrict__ pos, float* __restrict__ ehat,
                              int T, int L, int D) {
  long i = blockIdx.x * (long)blockDim.x + threadIdx.x;
  if (i >= (long)T * D) return;
  int t = (int)(i / D), d = (int)(i % D);
  ehat[i] = full[((size_t)seg[t] * L + pos[t]) * D + d];
}

// beta = tanh([hfin|ehat] @ w2^T) @ v3^T ; c = beta * hfin   (one block per row)
__global__ void k_beta_c(const float* __restrict__ hfin, const float* __restrict__ ehat,
                         const float* __restrict__ w2, const float* __restrict__ v3,
                         float* __restrict__ c, int D) {
  int t = blockIdx.x, d = threadIdx.x;
  const float* hr = hfin + (size_t)t * D;
  const float* er = ehat + (size_t)t * D;
  const float* wr = w2 + (size_t)d * 2 * D;
  float a = 0.f;
  for (int j = 0; j < D; ++j) a += hr[j] * wr[j];
  for (int j = 0; j < D; ++j) a += er[j] * wr[D + j];
  __shared__ float s[128];
  s[d] = tanhf(a) * v3[d];
  __syncthreads();
  for (int o = blockDim.x >> 1; o > 0; o >>= 1) {
    if (d < o) s[d] += s[d + o];
    __syncthreads();
  }
  c[(size_t)t * D + d] = s[0] * hr[d];
}

__global__ void k_zlong(const float* __restrict__ c, const int* __restrict__ seg,
                        float* __restrict__ zl, int T, int D) {
  long i = blockIdx.x * (long)blockDim.x + threadIdx.x;
  if (i >= (long)T * D) return;
  int t = (int)(i / D), d = (int)(i % D);
  atomicAdd(&zl[(size_t)seg[t] * D + d], c[i]);
}

__global__ void k_scatter_cpad(const float* __restrict__ c, const int* __restrict__ seg,
                               const int* __restrict__ pos, float* __restrict__ cpad,
                               int T, int L, int D) {
  long i = blockIdx.x * (long)blockDim.x + threadIdx.x;
  if (i >= (long)T * D) return;
  int t = (int)(i / D), d = (int)(i % D);
  cpad[((size_t)seg[t] * L + pos[t]) * D + d] = c[i];
}

// f = sigmoid((zl+zs).w3 + b3); z_final -> bf16 A operand (padded rows zeroed)
__global__ void k_fuse(const float* __restrict__ zl, const float* __restrict__ zs,
                       const float* __restrict__ w3, const float* __restrict__ b3,
                       unsigned short* __restrict__ Abf, int B, int D) {
  int b = blockIdx.x, d = threadIdx.x;
  __shared__ float s[128];
  if (b >= B) { Abf[(size_t)b * D + d] = 0; return; }
  float l = zl[(size_t)b * D + d];
  float sh = zs[(size_t)b * D + d];
  s[d] = (l + sh) * w3[d];
  __syncthreads();
  for (int o = blockDim.x >> 1; o > 0; o >>= 1) {
    if (d < o) s[d] += s[d + o];
    __syncthreads();
  }
  float f = sigm(s[0] + b3[0]);
  Abf[(size_t)b * D + d] = f2bf_u(f * l + (1.f - f) * sh);
}

__global__ void k_cvt_bf16(const float* __restrict__ src, unsigned short* __restrict__ dst,
                           long n) {
  long i = blockIdx.x * (long)blockDim.x + threadIdx.x;
  if (i < n) dst[i] = f2bf_u(src[i]);
}

// ---------------------------------------------------------------------------
// Host orchestration
// ---------------------------------------------------------------------------
extern "C" void kernel_launch(void* const* d_in, const int* in_sizes, int n_in,
                              void* d_out, int out_size, void* d_ws, size_t ws_size,
                              hipStream_t stream)
{
  const int*   items    = (const int*)d_in[0];
  const int*   edge     = (const int*)d_in[1];
  const float* ew       = (const float*)d_in[2];
  const int*   item2idx = (const int*)d_in[3];
  const int*   bins     = (const int*)d_in[4];
  const int*   seg      = (const int*)d_in[5];
  const int*   pos      = (const int*)d_in[6];
  const float* item_emb = (const float*)d_in[8];
  const float* bin_emb  = (const float*)d_in[9];
  const float* Wih_f = (const float*)d_in[10];
  const float* Whh_f = (const float*)d_in[11];
  const float* bih_f = (const float*)d_in[12];
  const float* bhh_f = (const float*)d_in[13];
  const float* Wih_b = (const float*)d_in[14];
  const float* Whh_b = (const float*)d_in[15];
  const float* bih_b = (const float*)d_in[16];
  const float* bhh_b = (const float*)d_in[17];
  const float* w1    = (const float*)d_in[18];
  const float* w2    = (const float*)d_in[19];
  const float* v3    = (const float*)d_in[20];
  const float* Wih_s = (const float*)d_in[21];
  const float* Whh_s = (const float*)d_in[22];
  const float* bih_s = (const float*)d_in[23];
  const float* bhh_s = (const float*)d_in[24];
  const float* w3    = (const float*)d_in[25];
  const float* b3    = (const float*)d_in[26];

  const int T  = in_sizes[0];
  const int E  = in_sizes[2];
  const int Bn = in_sizes[7];
  const int D  = in_sizes[20];          // 128 (template K below assumes this)
  const int L  = in_sizes[4] / Bn;      // 50
  const int NI = in_sizes[8] / D;       // 100000
  const int D3 = 3 * D;
  const int Bpad = (Bn + 15) & ~15;     // 1008
  const int LAYERS = 3;
  constexpr int KC = 128;               // compile-time K for WMMA kernels
  constexpr int NT_SMALL = 3;           // N-tiles/wave for N=384 GEMMs (384 % 48 == 0)
  constexpr int NT_FINAL = 5;           // N-tiles/wave for N=100000 (6250 % 5 == 0)

  // bump allocator over workspace
  char* base = (char*)d_ws;
  size_t off = 0;
  auto alloc = [&](size_t bytes) -> void* {
    off = (off + 255) & ~(size_t)255;
    void* p = base + off;
    off += bytes;
    return p;
  };
  float* acc  = (float*)alloc((size_t)T * D * 4);
  float* x    = (float*)alloc((size_t)T * D * 4);       // later reused as ehat
  float* xnew = (float*)alloc((size_t)T * D * 4);       // later reused as c
  float* hfin = (float*)alloc((size_t)T * D * 4);
  float* e    = (float*)alloc((size_t)Bn * L * D * 4);  // later reused as c_pad
  float* xp   = (float*)alloc((size_t)Bn * L * D3 * 4);
  float* outf = (float*)alloc((size_t)Bn * L * D * 4);  // later e_hat_full
  float* outb = (float*)alloc((size_t)Bn * L * D * 4);
  float* gh   = (float*)alloc((size_t)Bpad * D3 * 4);
  float* hcur = (float*)alloc((size_t)Bpad * D * 4);
  float* zlng = (float*)alloc((size_t)Bn * D * 4);
  unsigned short* Abf    = (unsigned short*)alloc((size_t)Bpad * D * 2);
  unsigned short* itembf = (unsigned short*)alloc((size_t)NI * D * 2);
  float* ehat = x;      // reuse (GCN ping buffers dead after k_hfinal)
  float* cbuf = xnew;
  float* cpad = e;      // reuse (e dead after both directional xp GEMMs)

  auto blocks = [](long n, int t) { return (unsigned)((n + t - 1) / t); };
  auto gemm = [&](const float* A, const float* W, const float* bias, float* C,
                  int M, int N, int revL) {
    int tileGroups = (M / 16) * (N / (16 * NT_SMALL));
    k_gemm_f32_wmma<KC, NT_SMALL><<<blocks(tileGroups, 4), 128, 0, stream>>>(
        A, W, bias, C, M, N, revL);
  };

  // ---- GCN over item graph --------------------------------------------------
  k_gather_items<<<blocks((long)T * D, 256), 256, 0, stream>>>(items, item_emb, acc, x, T, D);
  for (int l = 0; l < LAYERS; ++l) {
    k_zero<<<blocks((long)T * D, 256), 256, 0, stream>>>(xnew, (long)T * D);
    k_gcn_scatter<<<blocks((long)E * D, 256), 256, 0, stream>>>(edge, edge + E, ew, x, xnew, E, D);
    k_gcn_accum<<<blocks((long)T * D, 256), 256, 0, stream>>>(acc, x, xnew, (long)T * D);
  }
  k_hfinal<<<blocks((long)T * D, 256), 256, 0, stream>>>(acc, item2idx, hfin, T, D,
                                                         1.f / (LAYERS + 1));

  // ---- time-bin embeddings + bidirectional GRU ------------------------------
  k_bin_gather<<<blocks((long)Bn * L * D, 256), 256, 0, stream>>>(bins, bin_emb, e, Bn * L, D);

  // forward GRU
  k_zero<<<blocks((long)Bpad * D, 256), 256, 0, stream>>>(hcur, (long)Bpad * D);
  gemm(e, Wih_f, bih_f, xp, Bn * L, D3, 0);
  for (int t = 0; t < L; ++t) {
    gemm(hcur, Whh_f, bhh_f, gh, Bpad, D3, 0);
    k_gru_gate<<<blocks((long)Bn * D, 256), 256, 0, stream>>>(xp, gh, hcur, outf, t, t, Bn, L, D);
  }
  // backward GRU (input time-reversed via revL; outputs stored re-reversed)
  k_zero<<<blocks((long)Bpad * D, 256), 256, 0, stream>>>(hcur, (long)Bpad * D);
  gemm(e, Wih_b, bih_b, xp, Bn * L, D3, L);
  for (int t = 0; t < L; ++t) {
    gemm(hcur, Whh_b, bhh_b, gh, Bpad, D3, 0);
    k_gru_gate<<<blocks((long)Bn * D, 256), 256, 0, stream>>>(xp, gh, hcur, outb, t, L - 1 - t,
                                                              Bn, L, D);
  }

  // ---- gate fuse, ragged gather, attention, long-term interest --------------
  k_egate<<<(unsigned)(Bn * L), D, 0, stream>>>(outf, outb, w1, D);
  k_gather_ehat<<<blocks((long)T * D, 256), 256, 0, stream>>>(outf, seg, pos, ehat, T, L, D);
  k_beta_c<<<(unsigned)T, D, 0, stream>>>(hfin, ehat, w2, v3, cbuf, D);
  k_zero<<<blocks((long)Bn * D, 256), 256, 0, stream>>>(zlng, (long)Bn * D);
  k_zlong<<<blocks((long)T * D, 256), 256, 0, stream>>>(cbuf, seg, zlng, T, D);

  // ---- short GRU over right-aligned padded c --------------------------------
  k_zero<<<blocks((long)Bn * L * D, 256), 256, 0, stream>>>(cpad, (long)Bn * L * D);
  k_scatter_cpad<<<blocks((long)T * D, 256), 256, 0, stream>>>(cbuf, seg, pos, cpad, T, L, D);
  k_zero<<<blocks((long)Bpad * D, 256), 256, 0, stream>>>(hcur, (long)Bpad * D);
  gemm(cpad, Wih_s, bih_s, xp, Bn * L, D3, 0);
  for (int t = 0; t < L; ++t) {
    gemm(hcur, Whh_s, bhh_s, gh, Bpad, D3, 0);
    k_gru_gate<<<blocks((long)Bn * D, 256), 256, 0, stream>>>(xp, gh, hcur, nullptr, t, 0,
                                                              Bn, L, D);
  }

  // ---- fuse + final scoring GEMM (bf16 WMMA, output-BW bound) ---------------
  k_fuse<<<(unsigned)Bpad, D, 0, stream>>>(zlng, hcur, w3, b3, Abf, Bn, D);
  k_cvt_bf16<<<blocks((long)NI * D, 256), 256, 0, stream>>>(item_emb, itembf, (long)NI * D);
  {
    int tileGroups = (Bpad / 16) * (NI / (16 * NT_FINAL));
    k_gemm_bf16_wmma<KC, NT_FINAL><<<blocks(tileGroups, 4), 128, 0, stream>>>(
        Abf, itembf, (float*)d_out, Bpad, Bn, NI);
  }
}
